// SSAN_24988119728301
// MI455X (gfx1250) — matrix operand order repaired and checked by reading, not compiled, not run on hardware
//
#include <hip/hip_runtime.h>
#include <hip/hip_bf16.h>
#include <math.h>

#define D_MODEL 768
#define BATCH   8192
#define KNOW    4096
#define TOP_K   64

typedef __bf16 bf16;
typedef __attribute__((ext_vector_type(16))) __bf16 v16bf;
typedef __attribute__((ext_vector_type(8)))  __bf16 v8bf;
typedef __attribute__((ext_vector_type(8)))  float  v8f;

// ---------------------------------------------------------------------------
// Elementwise prep: mix = wa*a + wp*p (bf16), and bf16 cast of p.
// ---------------------------------------------------------------------------
__global__ __launch_bounds__(256) void mix_cast_kernel(
    const float* __restrict__ a, const float* __restrict__ p,
    bf16* __restrict__ mix_out, bf16* __restrict__ pe_out,
    int n, float wa, float wp)
{
  int i = blockIdx.x * blockDim.x + threadIdx.x;
  if (i < n) {
    mix_out[i] = (bf16)(a[i] * wa + p[i] * wp);
    pe_out[i]  = (bf16)p[i];
  }
}

__global__ __launch_bounds__(256) void cast_kernel(
    const float* __restrict__ w, bf16* __restrict__ o, int n)
{
  int i = blockIdx.x * blockDim.x + threadIdx.x;
  if (i < n) o[i] = (bf16)w[i];
}

// ---------------------------------------------------------------------------
// C[M,N] = A[M,K] * Bm[N,K]^T   (both row-major, K contiguous), bf16 in,
// f32 WMMA accumulate.
//
// Wave tile 64x64 = 4x4 v_wmma_f32_16x16x32_bf16 tiles -> 16 WMMAs per
// K-step of 32, fed by 8 A-fragment b128 loads + 8 B-fragment b128 loads
// (1.0 loads/WMMA).  Block = 8 waves arranged 2(M) x 4(N) -> 128 x 256
// block tile.  Grid: (N/256, M/128).  All problem dims divide evenly.
//
// MODE 0: outB = bf16(acc + resid[row,col])             (fused residual)
// MODE 1: outF = acc * scale                             (pe_sims)
// MODE 2: outF = (pe[row,col] < thr[row]) ? 0 : acc*scale, nontemporal store
// ---------------------------------------------------------------------------
template <int MODE>
__global__ __launch_bounds__(256) void gemm_nt_wmma(
    const bf16* __restrict__ A, const bf16* __restrict__ Bm,
    int M, int N, int K, float scale,
    const bf16* __restrict__ resid,
    const float* __restrict__ pe,
    const float* __restrict__ thr,
    float* __restrict__ outF, bf16* __restrict__ outB)
{
  const int tid  = threadIdx.x;
  const int lane = tid & 31;
  const int w    = tid >> 5;        // wave id 0..7
  const int wm   = w >> 2;          // 0..1  (M direction)
  const int wn   = w & 3;           // 0..3  (N direction)
  const int r    = lane & 15;
  const int hi   = lane >> 4;       // 0/1: lane half

  const int m0 = blockIdx.y * 128 + wm * 64;
  const int n0 = blockIdx.x * 256 + wn * 64;

  v8f acc[4][4];
#pragma unroll
  for (int i = 0; i < 4; i++)
#pragma unroll
    for (int j = 0; j < 4; j++) { v8f z = {}; acc[i][j] = z; }

  // A fragment (16x32 bf16): lane r holds row; lanes 0-15 cover K=0..15,
  // lanes 16-31 cover K=16..31 (same contiguous order as B).
  const bf16* pa[4];
  // B fragment (32x16 bf16): lane r holds col; lanes 0-15 K=0..15,
  // lanes 16-31 K=16..31 -> one contiguous 32B chunk per lane.
  const bf16* pb[4];
#pragma unroll
  for (int i = 0; i < 4; i++) {
    pa[i] = A  + (size_t)(m0 + i * 16 + r) * K + hi * 16;
    pb[i] = Bm + (size_t)(n0 + i * 16 + r) * K + hi * 16;
  }

  for (int k0 = 0; k0 < K; k0 += 32) {
    v16bf a[4], b[4];
#pragma unroll
    for (int i = 0; i < 4; i++) {
      v8bf lo = *(const v8bf*)(pa[i] + k0);
      v8bf hh = *(const v8bf*)(pa[i] + k0 + 8);
#pragma unroll
      for (int t = 0; t < 8; t++) { a[i][t] = lo[t]; a[i][t + 8] = hh[t]; }
    }
#pragma unroll
    for (int j = 0; j < 4; j++) b[j] = *(const v16bf*)(pb[j] + k0);

    // Unconditional speculative prefetch of the next K tile (reads past the
    // end of the buffer are silently dropped); locality 3 -> near cache.
#pragma unroll
    for (int i = 0; i < 4; i++) {
      __builtin_prefetch(pa[i] + k0 + 32, 0, 3);
      __builtin_prefetch(pb[i] + k0 + 32, 0, 3);
    }

#pragma unroll
    for (int i = 0; i < 4; i++)
#pragma unroll
      for (int j = 0; j < 4; j++)
        acc[i][j] = __builtin_amdgcn_wmma_f32_16x16x32_bf16(
            false, a[i], false, b[j], (short)0, acc[i][j], false, false);
  }

  // D layout (f32 16x16): lanes 0-15 col=lane rows 0..7; lanes 16-31 rows 8..15.
#pragma unroll
  for (int i = 0; i < 4; i++) {
    const int mbase = m0 + i * 16 + hi * 8;
#pragma unroll
    for (int j = 0; j < 4; j++) {
      const int ncol = n0 + j * 16 + r;
#pragma unroll
      for (int t = 0; t < 8; t++) {
        const size_t idx = (size_t)(mbase + t) * N + ncol;
        const float v = acc[i][j][t];
        if (MODE == 0) {
          outB[idx] = (bf16)(v + (float)resid[idx]);
        } else if (MODE == 1) {
          outF[idx] = v * scale;
        } else {
          const float p  = pe[idx];
          const float t0 = thr[mbase + t];
          __builtin_nontemporal_store((p < t0) ? 0.0f : v * scale, &outF[idx]);
        }
      }
    }
  }
}

// ---------------------------------------------------------------------------
// Per-row 64th-largest value of pe_sims[row, 0..N): radix binary search on
// order-preserving uint keys.  One block per row; row cached in LDS (16KB).
// Deterministic, no sorting.
// ---------------------------------------------------------------------------
__global__ __launch_bounds__(128) void topk_thresh_kernel(
    const float* __restrict__ pe, float* __restrict__ thr, int N, int k)
{
  __shared__ unsigned keys[KNOW];
  __shared__ int cnt;
  const int row = blockIdx.x;
  const int tid = threadIdx.x;
  const float* prow = pe + (size_t)row * N;

  for (int i = tid; i < N; i += 128) {
    unsigned u = __float_as_uint(prow[i]);
    keys[i] = (u & 0x80000000u) ? ~u : (u | 0x80000000u);  // order-preserving
  }
  __syncthreads();

  unsigned res = 0;
  for (int bit = 31; bit >= 0; --bit) {
    const unsigned cand = res | (1u << bit);
    if (tid == 0) cnt = 0;
    __syncthreads();
    int local = 0;
    for (int i = tid; i < N; i += 128) local += (keys[i] >= cand) ? 1 : 0;
    atomicAdd(&cnt, local);
    __syncthreads();
    if (cnt >= k) res = cand;   // keep bit if still >= k elements above
    __syncthreads();
  }
  if (tid == 0) {
    thr[row] = (res & 0x80000000u) ? __uint_as_float(res & 0x7fffffffu)
                                   : __uint_as_float(~res);
  }
}

// ---------------------------------------------------------------------------
extern "C" void kernel_launch(void* const* d_in, const int* in_sizes, int n_in,
                              void* d_out, int out_size, void* d_ws, size_t ws_size,
                              hipStream_t stream)
{
  (void)in_sizes; (void)n_in; (void)out_size; (void)ws_size;
  const float* ae_q  = (const float*)d_in[0];
  const float* ae_kv = (const float*)d_in[1];
  const float* pe_q  = (const float*)d_in[2];
  const float* pe_kv = (const float*)d_in[3];
  const float* Wq    = (const float*)d_in[4];
  const float* Wk    = (const float*)d_in[5];
  float* out = (float*)d_out;

  const float inv_sqrt_d = 1.0f / sqrtf((float)D_MODEL);
  const float pe_w = 0.5f, a_scale = 0.5f;  // PE_WEIGHT, |1 - PE_WEIGHT|

  // -------- workspace layout (256B aligned slabs) --------
  char* base = (char*)d_ws;
  size_t off = 0;
  auto carve = [&](size_t bytes) -> char* {
    char* p = base + off;
    off = (off + bytes + 255) & ~(size_t)255;
    return p;
  };
  bf16* qin_b   = (bf16*)carve((size_t)BATCH * D_MODEL * 2);
  bf16* peq_b   = (bf16*)carve((size_t)BATCH * D_MODEL * 2);
  bf16* kin_b   = (bf16*)carve((size_t)KNOW  * D_MODEL * 2);
  bf16* pekv_b  = (bf16*)carve((size_t)KNOW  * D_MODEL * 2);
  bf16* wq_b    = (bf16*)carve((size_t)D_MODEL * D_MODEL * 2);
  bf16* wk_b    = (bf16*)carve((size_t)D_MODEL * D_MODEL * 2);
  bf16* query_b = (bf16*)carve((size_t)BATCH * D_MODEL * 2);
  bf16* key_b   = (bf16*)carve((size_t)KNOW  * D_MODEL * 2);
  float* pesims = (float*)carve((size_t)BATCH * KNOW * 4);   // L2-resident (134MB < 192MB)
  float* thresh = (float*)carve((size_t)BATCH * 4);

  // -------- 1) elementwise mixes + bf16 casts --------
  {
    int n = BATCH * D_MODEL;
    mix_cast_kernel<<<(n + 255) / 256, 256, 0, stream>>>(ae_q, pe_q, qin_b, peq_b, n, a_scale, pe_w);
  }
  {
    int n = KNOW * D_MODEL;
    mix_cast_kernel<<<(n + 255) / 256, 256, 0, stream>>>(ae_kv, pe_kv, kin_b, pekv_b, n, a_scale, pe_w);
  }
  {
    int n = D_MODEL * D_MODEL;
    cast_kernel<<<(n + 255) / 256, 256, 0, stream>>>(Wq, wq_b, n);
    cast_kernel<<<(n + 255) / 256, 256, 0, stream>>>(Wk, wk_b, n);
  }

  // -------- 2) query = q_in @ Wq^T + q_in ; key = k_in @ Wk^T + k_in --------
  {
    dim3 g(D_MODEL / 256, BATCH / 128);
    gemm_nt_wmma<0><<<g, 256, 0, stream>>>(qin_b, wq_b, BATCH, D_MODEL, D_MODEL,
                                           1.0f, qin_b, nullptr, nullptr, nullptr, query_b);
  }
  {
    dim3 g(D_MODEL / 256, KNOW / 128);
    gemm_nt_wmma<0><<<g, 256, 0, stream>>>(kin_b, wk_b, KNOW, D_MODEL, D_MODEL,
                                           1.0f, kin_b, nullptr, nullptr, nullptr, key_b);
  }

  // -------- 3) pe_sims = pe_q @ pe_kv^T * inv_sqrt_d --------
  {
    dim3 g(KNOW / 256, BATCH / 128);
    gemm_nt_wmma<1><<<g, 256, 0, stream>>>(peq_b, pekv_b, BATCH, KNOW, D_MODEL,
                                           inv_sqrt_d, nullptr, nullptr, nullptr, pesims, nullptr);
  }

  // -------- 4) per-row 64th-largest pe_sim --------
  topk_thresh_kernel<<<BATCH, 128, 0, stream>>>(pesims, thresh, KNOW, TOP_K);

  // -------- 5) att = (query @ key^T) * inv_sqrt_d, masked, single write --------
  {
    dim3 g(KNOW / 256, BATCH / 128);
    gemm_nt_wmma<2><<<g, 256, 0, stream>>>(query_b, key_b, BATCH, KNOW, D_MODEL,
                                           inv_sqrt_d, nullptr, pesims, thresh, out, nullptr);
  }
}
